// SparseTriangleMultiplicativeUpdate_90383291777390
// MI455X (gfx1250) — matrix operand order repaired and checked by reading, not compiled, not run on hardware
//
#include <hip/hip_runtime.h>

// Sparse triangle multiplicative update, fused one-block-per-node, WMMA f16.
//
// N=2000 nodes, K=16 neighbors, C_S=128, C_Z=64, NUM_RBF=64.
// Factorization: pair@W_gate = G1[i] + G2[j] with G1=nf@Wg[:128], G2=nf@Wg[128:].
// All GEMMs run on v_wmma_f32_16x16x32_f16 (f32 accum).
// Weights pre-transposed/converted to f16 once per launch into d_ws so each
// WMMA B fragment is two global_load_b128 per lane; phase-4 B fragments and
// G2 gathers are hoisted (loop-invariant per wave), Sigma_i accumulated in
// registers with one ds_add_f32 burst at the end.

#define NN    2000
#define KK    16
#define CS    128
#define CZ    64
#define NRBF  64

// f16-packed transposed weights in d_ws (halves):
//   wgT [64][256] at OFF_WG,  wdbT[64][64] at OFF_DB,
//   weT [64][64]  at OFF_WE,  woT [64][64] at OFF_WO
#define OFF_WG 0
#define OFF_DB 16384
#define OFF_WE 20480
#define OFF_WO 24576
#define WPK_HALVES 28672   // * 2 bytes = 57344 B workspace needed

typedef __attribute__((ext_vector_type(16))) _Float16 v16h;
typedef __attribute__((ext_vector_type(8)))  _Float16 v8h;
typedef __attribute__((ext_vector_type(4)))  _Float16 v4h;
typedef __attribute__((ext_vector_type(8)))  float    v8f;
typedef __attribute__((ext_vector_type(4)))  float    v4f;

// ---------------------------------------------------------------------------
// Prep kernel: transpose + f32->f16 convert all weight matrices into d_ws.
// wT[n][k] = W[k][n].  One thread per output half.
// ---------------------------------------------------------------------------
__global__ __launch_bounds__(256)
void pack_weights_kernel(const float* __restrict__ w_gate,  // [256][64]
                         const float* __restrict__ w_db,    // [64][64]
                         const float* __restrict__ w_edge,  // [64][64]
                         const float* __restrict__ w_out,   // [64][64]
                         _Float16* __restrict__ wpk)
{
    int t = blockIdx.x * 256 + threadIdx.x;
    if (t < OFF_DB) {                       // wgT [64][256]
        int n = t >> 8, k = t & 255;
        wpk[t] = (_Float16)w_gate[k * CZ + n];
    } else if (t < OFF_WE) {                // wdbT [64][64]
        int u = t - OFF_DB;
        int n = u >> 6, k = u & 63;
        wpk[t] = (_Float16)w_db[k * CZ + n];
    } else if (t < OFF_WO) {                // weT [64][64]
        int u = t - OFF_WE;
        int n = u >> 6, k = u & 63;
        wpk[t] = (_Float16)w_edge[k * CZ + n];
    } else if (t < WPK_HALVES) {            // woT [64][64]
        int u = t - OFF_WO;
        int n = u >> 6, k = u & 63;
        wpk[t] = (_Float16)w_out[k * CZ + n];
    }
}

// ---------------------------------------------------------------------------
// Fragment loaders
// ---------------------------------------------------------------------------

// A fragment (16x32 f16) from row-major f16 LDS array, row stride RS halves.
// Layout used here (contiguous k, as in B): lane m=lane&15 is matrix row; lanes<16 carry K = k0+{0..15},
// lanes>=16 carry K = k0+{16..31}.  Two aligned 16B LDS loads per lane.
__device__ __forceinline__ v16h load_a_lds(const _Float16* S, int row_stride,
                                           int row0, int k0, int lane) {
    int m  = lane & 15;
    int hi = (lane >> 4) << 4;                 // 0 or 16
    const _Float16* p = S + (row0 + m) * row_stride + k0 + hi;
    v8h lo = *(const v8h*)p;                   // K = k0+hi    .. +7
    v8h hv = *(const v8h*)(p + 8);             // K = k0+hi+8  .. +15
    v16h a;
#pragma unroll
    for (int e = 0; e < 8; ++e) { a[e] = lo[e]; a[e + 8] = hv[e]; }
    return a;
}

// B fragment (32x16 f16) from f16-packed transposed weights WT[n][k] (stride ks).
// Lane carries column n = n0+(lane&15); K contiguous -> two global_load_b128.
__device__ __forceinline__ v16h load_b_pk(const _Float16* __restrict__ WT,
                                          int kstride, int k0, int n0, int lane) {
    int hi = (lane >> 4) << 4;                 // 0 or 16
    const _Float16* p = WT + (n0 + (lane & 15)) * kstride + k0 + hi;
    v8h lo = *(const v8h*)p;                   // K = k0+hi .. +7
    v8h hv = *(const v8h*)(p + 8);             // K = k0+hi+8 .. +15
    v16h b;
#pragma unroll
    for (int e = 0; e < 8; ++e) { b[e] = lo[e]; b[e + 8] = hv[e]; }
    return b;
}

// Fallback: B fragment from row-major f32 weight W[k][n] (n-stride CZ).
__device__ __forceinline__ v16h load_b_glb(const float* __restrict__ W,
                                           int krow0, int n0, int lane) {
    int hi  = (lane >> 4) << 4;
    int col = n0 + (lane & 15);
    v16h b;
#pragma unroll
    for (int e = 0; e < 16; ++e)
        b[e] = (_Float16)W[(krow0 + hi + e) * CZ + col];
    return b;
}

// ---------------------------------------------------------------------------
// Main fused kernel
// ---------------------------------------------------------------------------
template <bool USE_PK>
__global__ __launch_bounds__(256)
void stmu_kernel(const float* __restrict__ node_features,   // [N, CS]
                 const float* __restrict__ trans,           // [N, 3]
                 const float* __restrict__ edge_features,   // [N*KK, CZ]
                 const int*   __restrict__ edge_index,      // [2, N*KK] (row 0 = src)
                 const float* __restrict__ w_gate,          // [2*CS, CZ]
                 const float* __restrict__ b_gate,          // [CZ]
                 const float* __restrict__ w_db,            // [NRBF, CZ]
                 const float* __restrict__ b_db,            // [CZ]
                 const float* __restrict__ w_edge,          // [CZ, CZ]
                 const float* __restrict__ b_edge,          // [CZ]
                 const float* __restrict__ ln_g,            // [CZ]
                 const float* __restrict__ ln_b,            // [CZ]
                 const float* __restrict__ w_out,           // [CZ, CZ]
                 const float* __restrict__ b_out,           // [CZ]
                 const _Float16* __restrict__ wpk,          // packed weights
                 float* __restrict__ out)                   // [N*KK, CZ]
{
    __shared__ int   s_idx[KK];
    __shared__ float s_t[KK][4];
    __shared__ __align__(16) _Float16 s_nf [KK][CS];        // 4 KB  (A: gate GEMMs)
    __shared__ __align__(16) _Float16 s_kfh[KK][CZ];        // 2 KB  (A: kf GEMM)
    __shared__ __align__(16) _Float16 s_rbf[KK * KK][NRBF]; // 32 KB (A: DB GEMM)
    __shared__ float s_G[2][KK][CZ];                        // 8 KB  (G1, G2)
    __shared__ float s_kfp[KK][CZ];                         // 4 KB  (kf projected)
    __shared__ __align__(16) float s_upd[KK][CZ];           // 4 KB  (i-reduction)
    __shared__ __align__(16) _Float16 s_updh[KK][CZ];       // 2 KB  (A: out GEMM)

    const int n    = blockIdx.x;
    const int tid  = threadIdx.x;
    const int lane = tid & 31;
    const int wv   = tid >> 5;                 // wave id, 8 waves of 32

    if (tid < KK) s_idx[tid] = edge_index[n * KK + tid];    // src nodes
    __syncthreads();

    // ---- phase 1: gather node/edge features -> f16 LDS (float4 loads) ----
#pragma unroll
    for (int v = 0; v < 2; ++v) {              // 16*128 floats = 512 float4
        int t  = tid + v * 256;
        int r  = t >> 5, c4 = t & 31;
        v4f x = *(const v4f*)(node_features + s_idx[r] * CS + c4 * 4);
        v4h h;
#pragma unroll
        for (int q = 0; q < 4; ++q) h[q] = (_Float16)x[q];
        *(v4h*)&s_nf[r][c4 * 4] = h;
    }
    {                                          // 16*64 floats = 256 float4
        int r  = tid >> 4, c4 = tid & 15;
        v4f x = *(const v4f*)(edge_features + (n * KK + r) * CZ + c4 * 4);
        v4h h;
#pragma unroll
        for (int q = 0; q < 4; ++q) h[q] = (_Float16)x[q];
        *(v4h*)&s_kfh[r][c4 * 4] = h;
        v4f z = {};
        *(v4f*)&s_upd[r][c4 * 4] = z;          // zero the i-reduction buffer
    }
    if (tid < KK * 3) {
        int r = tid / 3, c = tid % 3;
        s_t[r][c] = trans[s_idx[r] * 3 + c];
    }
    __syncthreads();

    // ---- phase 2: pair distances + RBF expansion (one thread per (i,j)) ----
    {
        int i = tid >> 4, j = tid & 15;
        float dx = s_t[i][0] - s_t[j][0] + 1e-8f;
        float dy = s_t[i][1] - s_t[j][1] + 1e-8f;
        float dz = s_t[i][2] - s_t[j][2] + 1e-8f;
        float d  = sqrtf(dx * dx + dy * dy + dz * dz);
        const float inv_sigma = (float)NRBF / 20.0f;         // sigma = 20/64
        for (int r = 0; r < NRBF; ++r) {
            float mu = 20.0f * (float)r / (float)(NRBF - 1); // linspace(0,20,64)
            float u  = (d - mu) * inv_sigma;
            s_rbf[tid][r] = (_Float16)__expf(-u * u);
        }
    }
    __syncthreads();

    // ---- phase 3: G1/G2 = nf @ W_gate halves (8 tiles), kf @ W_edge (4 tiles) ----
    for (int job = wv; job < 12; job += 8) {
        v8f acc = {};
        int hiC = (lane >> 4) << 3;            // 0 or 8 (C layout row offset)
        if (job < 8) {
            int g = job >> 2, ct = job & 3;    // g: top/bottom half of W_gate
#pragma unroll
            for (int ks = 0; ks < 4; ++ks) {   // K = 128
                v16h a = load_a_lds(&s_nf[0][0], CS, 0, ks * 32, lane);
                v16h b = USE_PK
                    ? load_b_pk(wpk + OFF_WG, 2 * CS, g * CS + ks * 32, ct * 16, lane)
                    : load_b_glb(w_gate, g * CS + ks * 32, ct * 16, lane);
                acc = __builtin_amdgcn_wmma_f32_16x16x32_f16(
                          false, a, false, b, (short)0, acc, false, false);
            }
            int h = ct * 16 + (lane & 15);
#pragma unroll
            for (int r = 0; r < 8; ++r)
                s_G[g][r + hiC][h] = acc[r];
        } else {
            int ct = job - 8;
#pragma unroll
            for (int ks = 0; ks < 2; ++ks) {   // K = 64
                v16h a = load_a_lds(&s_kfh[0][0], CZ, 0, ks * 32, lane);
                v16h b = USE_PK
                    ? load_b_pk(wpk + OFF_WE, CZ, ks * 32, ct * 16, lane)
                    : load_b_glb(w_edge, ks * 32, ct * 16, lane);
                acc = __builtin_amdgcn_wmma_f32_16x16x32_f16(
                          false, a, false, b, (short)0, acc, false, false);
            }
            int h  = ct * 16 + (lane & 15);
            float be = b_edge[h];
#pragma unroll
            for (int r = 0; r < 8; ++r)
                s_kfp[r + hiC][h] = acc[r] + be;
        }
    }
    __syncthreads();

    // ---- phase 4: DB = RBF @ W_db, fused gate/scale + Sigma_i reduction ----
    // Wave handles ct = wv&3 (constant!) and i in {wv>>2 + 2s, s=0..7}.
    // B fragments and the G2 row gather are loop-invariant -> hoisted.
    // Sigma_i partials kept in registers; one ds_add_f32 burst at the end.
    {
        const int ct    = wv & 3;
        const int ibase = wv >> 2;
        const int hiC   = (lane >> 4) << 3;
        const int h     = ct * 16 + (lane & 15);

        v16h b0 = USE_PK ? load_b_pk(wpk + OFF_DB, CZ,  0, ct * 16, lane)
                         : load_b_glb(w_db,  0, ct * 16, lane);
        v16h b1 = USE_PK ? load_b_pk(wpk + OFF_DB, CZ, 32, ct * 16, lane)
                         : load_b_glb(w_db, 32, ct * 16, lane);

        float bg = b_gate[h];
        float bd = b_db[h];
        float g2[8];
#pragma unroll
        for (int r = 0; r < 8; ++r) g2[r] = s_G[1][r + hiC][h];

        float uacc[8];
#pragma unroll
        for (int r = 0; r < 8; ++r) uacc[r] = 0.0f;

#pragma unroll
        for (int s = 0; s < 8; ++s) {
            int i = ibase + 2 * s;
            v8f acc = {};
            v16h a0 = load_a_lds(&s_rbf[0][0], NRBF, i * 16,  0, lane);
            acc = __builtin_amdgcn_wmma_f32_16x16x32_f16(
                      false, a0, false, b0, (short)0, acc, false, false);
            v16h a1 = load_a_lds(&s_rbf[0][0], NRBF, i * 16, 32, lane);
            acc = __builtin_amdgcn_wmma_f32_16x16x32_f16(
                      false, a1, false, b1, (short)0, acc, false, false);

            float g1  = s_G[0][i][h] + bg;
            float kfi = s_kfp[i][h];
#pragma unroll
            for (int r = 0; r < 8; ++r) {
                float x    = g1 + g2[r];
                float gate = 1.0f / (1.0f + __expf(-x));
                uacc[r] += gate * (acc[r] + bd) * kfi;
            }
        }
#pragma unroll
        for (int r = 0; r < 8; ++r)
            atomicAdd(&s_upd[r + hiC][h], uacc[r]);   // ds_add_f32, 2-way contention
    }
    __syncthreads();

    // ---- phase 5: layernorm over h per row j ----
    if (tid < KK) {
        float m = 0.f;
        for (int h = 0; h < CZ; ++h) m += s_upd[tid][h];
        m *= (1.0f / CZ);
        float v = 0.f;
        for (int h = 0; h < CZ; ++h) { float d = s_upd[tid][h] - m; v += d * d; }
        v *= (1.0f / CZ);
        float rs = rsqrtf(v + 1e-5f);
        for (int h = 0; h < CZ; ++h)
            s_updh[tid][h] = (_Float16)((s_upd[tid][h] - m) * rs * ln_g[h] + ln_b[h]);
    }
    __syncthreads();

    // ---- phase 6: out = LN(upd) @ W_out + b_out, streamed to global ----
    if (wv < 4) {
        int ct = wv;
        v8f acc = {};
#pragma unroll
        for (int ks = 0; ks < 2; ++ks) {
            v16h a = load_a_lds(&s_updh[0][0], CZ, 0, ks * 32, lane);
            v16h b = USE_PK
                ? load_b_pk(wpk + OFF_WO, CZ, ks * 32, ct * 16, lane)
                : load_b_glb(w_out, ks * 32, ct * 16, lane);
            acc = __builtin_amdgcn_wmma_f32_16x16x32_f16(
                      false, a, false, b, (short)0, acc, false, false);
        }
        int hiC = (lane >> 4) << 3;
        int h   = ct * 16 + (lane & 15);
        float bo = b_out[h];
#pragma unroll
        for (int r = 0; r < 8; ++r)
            out[(n * KK + r + hiC) * CZ + h] = acc[r] + bo;
    }
}

extern "C" void kernel_launch(void* const* d_in, const int* in_sizes, int n_in,
                              void* d_out, int out_size, void* d_ws, size_t ws_size,
                              hipStream_t stream) {
    (void)in_sizes; (void)n_in; (void)out_size;
    const float* node_features = (const float*)d_in[0];
    const float* trans_p       = (const float*)d_in[1];
    const float* edge_features = (const float*)d_in[2];
    const int*   edge_index    = (const int*)d_in[3];
    const float* w_gate = (const float*)d_in[4];
    const float* b_gate = (const float*)d_in[5];
    const float* w_db   = (const float*)d_in[6];
    const float* b_db   = (const float*)d_in[7];
    const float* w_edge = (const float*)d_in[8];
    const float* b_edge = (const float*)d_in[9];
    const float* ln_g   = (const float*)d_in[10];
    const float* ln_b   = (const float*)d_in[11];
    const float* w_out  = (const float*)d_in[12];
    const float* b_out  = (const float*)d_in[13];

    bool use_pk = (ws_size >= (size_t)WPK_HALVES * sizeof(_Float16)) && (d_ws != nullptr);
    _Float16* wpk = (_Float16*)d_ws;

    if (use_pk) {
        pack_weights_kernel<<<(WPK_HALVES + 255) / 256, 256, 0, stream>>>(
            w_gate, w_db, w_edge, w_out, wpk);
        stmu_kernel<true><<<NN, 256, 0, stream>>>(
            node_features, trans_p, edge_features, edge_index,
            w_gate, b_gate, w_db, b_db, w_edge, b_edge,
            ln_g, ln_b, w_out, b_out, wpk, (float*)d_out);
    } else {
        stmu_kernel<false><<<NN, 256, 0, stream>>>(
            node_features, trans_p, edge_features, edge_index,
            w_gate, b_gate, w_db, b_db, w_edge, b_edge,
            ln_g, ln_b, w_out, b_out, wpk, (float*)d_out);
    }
}
